// _fasterRCNN_51754355917513
// MI455X (gfx1250) — matrix operand order repaired and checked by reading, not compiled, not run on hardware
//
#include <hip/hip_runtime.h>
#include <hip/hip_bf16.h>

// ---------------------------------------------------------------------------
// Faster-RCNN relation head on gfx1250 (MI455X), bf16 WMMA GEMM core.
// Register-blocked 64x32 per wave (4x2 WMMA tiles) + ping-pong K pipeline.
// ---------------------------------------------------------------------------

typedef __attribute__((ext_vector_type(16))) __bf16 bf16x16;
typedef __attribute__((ext_vector_type(8)))  __bf16 bf16x8;
typedef __attribute__((ext_vector_type(8)))  float  f32x8;

#define N_ROIS     2048
#define D_FEAT     4096
#define H_DIM      1024
#define NCLS       21
#define NBB        84          // 4 * NCLS
#define NCLS_PAD   32
#define NBB_PAD    96
#define MASK_WORDS (N_ROIS / 32)
#define IOU_THR    0.8f
#define PART_LIM   11          // PART_SIZE + 1

#define BLK_M      4           // 16-row subtiles per wave (64 rows)
#define BLK_N      2           // 16-col subtiles per wave (32 cols)

enum { OUT_BF16_ROW = 0, OUT_BF16_TRANS = 1, OUT_F32_ROW = 2 };

static __device__ __forceinline__ __bf16 f2bf(float f) { return (__bf16)f; }

// ---------------------------------------------------------------------------
// IoU mask: ov = (iou >= 0.8) & ~eye, cs = cumsum(ov, axis=1),
// mask = ov & (cs <= 11). One thread per row, boxes staged in LDS (32 KB).
// ---------------------------------------------------------------------------
__global__ __launch_bounds__(256)
void iou_mask_kernel(const float* __restrict__ rois, unsigned* __restrict__ maskb)
{
    __shared__ float4 sbox[N_ROIS];
    for (int i = threadIdx.x; i < N_ROIS; i += 256)
        sbox[i] = make_float4(rois[i * 5 + 1], rois[i * 5 + 2],
                              rois[i * 5 + 3], rois[i * 5 + 4]);
    __syncthreads();

    int row = blockIdx.x * 256 + threadIdx.x;
    if (row >= N_ROIS) return;
    float4 bi = sbox[row];
    float areai = (bi.z - bi.x + 1.f) * (bi.w - bi.y + 1.f);
    int cnt = 0;
    unsigned word = 0;
    for (int j = 0; j < N_ROIS; ++j) {
        float4 bj = sbox[j];
        float iw = fmaxf(fminf(bi.z, bj.z) - fmaxf(bi.x, bj.x) + 1.f, 0.f);
        float ih = fmaxf(fminf(bi.w, bj.w) - fmaxf(bi.y, bj.y) + 1.f, 0.f);
        float inter = iw * ih;
        float areaj = (bj.z - bj.x + 1.f) * (bj.w - bj.y + 1.f);
        float iou = inter / (areai + areaj - inter);
        bool ov = (iou >= IOU_THR) && (j != row);
        if (ov) {
            ++cnt;
            if (cnt <= PART_LIM) word |= (1u << (j & 31));
        }
        if ((j & 31) == 31) {
            maskb[(size_t)row * MASK_WORDS + (j >> 5)] = word;
            word = 0;
        }
    }
}

// ---------------------------------------------------------------------------
// Row L2 norms of pooled_feat; emit bf16 copies: pooled_bf and fn = x/||x||.
// ---------------------------------------------------------------------------
__global__ __launch_bounds__(256)
void norm_convert_kernel(const float* __restrict__ pooled,
                         __bf16* __restrict__ pooled_bf,
                         __bf16* __restrict__ fn_bf)
{
    __shared__ float red[256];
    int row = blockIdx.x;
    const float* p = pooled + (size_t)row * D_FEAT;
    float s = 0.f;
    for (int i = threadIdx.x; i < D_FEAT; i += 256) { float x = p[i]; s += x * x; }
    red[threadIdx.x] = s;
    __syncthreads();
    for (int off = 128; off > 0; off >>= 1) {
        if (threadIdx.x < off) red[threadIdx.x] += red[threadIdx.x + off];
        __syncthreads();
    }
    float inv = 1.f / fmaxf(sqrtf(red[0]), 1e-6f);
    for (int i = threadIdx.x; i < D_FEAT; i += 256) {
        float x = p[i];
        pooled_bf[(size_t)row * D_FEAT + i] = f2bf(x);
        fn_bf[(size_t)row * D_FEAT + i]     = f2bf(x * inv);
    }
}

// ---------------------------------------------------------------------------
// Transpose + convert f32[K][Nsrc] -> bf16 BT[Npad][K] (zero-padded cols).
// ---------------------------------------------------------------------------
__global__ __launch_bounds__(256)
void transpose_convert_kernel(const float* __restrict__ in, __bf16* __restrict__ out,
                              int K, int Nsrc, int Npad)
{
    size_t total = (size_t)Npad * K;
    for (size_t idx = (size_t)blockIdx.x * 256 + threadIdx.x; idx < total;
         idx += (size_t)gridDim.x * 256) {
        int n = (int)(idx / K);
        int k = (int)(idx % K);
        float v = (n < Nsrc) ? in[(size_t)k * Nsrc + n] : 0.f;
        out[idx] = f2bf(v);
    }
}

// ---------------------------------------------------------------------------
// bf16 WMMA GEMM: C[M,N] = A[M,K] * B[K,N]  (B supplied transposed, BT[N][K]).
// One wave = 64x32 output block (BLK_M x BLK_N 16x16 tiles).
// K stepped by 64 as a ping-pong pipeline over two fragment sets (no register
// rotation copies).  Requires K % 64 == 0 (true for all call sites here).
//
// A-frag (k order chosen here, not AMD's table order; B uses the same): lane l<16 -> row l, K {0..15};
// lane l>=16 -> row l-16, K {16..31}  => two contiguous 16B loads.
// B-frag: lane holds column (l&15), 16 consecutive K (pair-packed) starting at
// (l<16 ? 0 : 16) => one contiguous 32B load per BT row.  C-frag: VGPR r holds
// row (r + (l<16?0:8)), col (l&15).
// ---------------------------------------------------------------------------
__global__ __launch_bounds__(256)
void gemm_wmma_bf16(const __bf16* __restrict__ A, int lda,
                    const __bf16* __restrict__ BT, int ldbt,
                    void* __restrict__ out, int ldout,
                    int M, int Nlog, int Npad, int K,
                    const float* __restrict__ bias, int doRelu,
                    const unsigned* __restrict__ maskb, int outMode)
{
    const int tilesN = Npad / (16 * BLK_N);
    const int tilesM = M / (16 * BLK_M);
    const int tile = blockIdx.x * (blockDim.x >> 5) + (threadIdx.x >> 5);
    if (tile >= tilesM * tilesN) return;           // uniform per wave
    const int tM = tile / tilesN;
    const int tN = tile % tilesN;
    const int lane = threadIdx.x & 31;
    const int half = lane >> 4;
    const int l16  = lane & 15;
    const int m0 = tM * 16 * BLK_M;
    const int n0 = tN * 16 * BLK_N;

    const __bf16* aRow[BLK_M];
    const __bf16* bRow[BLK_N];
#pragma unroll
    for (int i = 0; i < BLK_M; ++i)
        aRow[i] = A + (size_t)(m0 + 16 * i + l16) * lda + half * 16;
#pragma unroll
    for (int j = 0; j < BLK_N; ++j)
        bRow[j] = BT + (size_t)(n0 + 16 * j + l16) * ldbt + half * 16;

    auto loadFrags = [&](int kk, bf16x16* a, bf16x16* b) {
#pragma unroll
        for (int i = 0; i < BLK_M; ++i) {
            const bf16x8* pa = (const bf16x8*)(aRow[i] + kk);
            bf16x8 a0 = pa[0];
            bf16x8 a1 = pa[1];                      // +8 elements
            a[i] = __builtin_shufflevector(a0, a1,
                       0, 1, 2, 3, 4, 5, 6, 7, 8, 9, 10, 11, 12, 13, 14, 15);
        }
#pragma unroll
        for (int j = 0; j < BLK_N; ++j)
            b[j] = *(const bf16x16*)(bRow[j] + kk);
    };

    f32x8 acc[BLK_M][BLK_N];
#pragma unroll
    for (int i = 0; i < BLK_M; ++i)
#pragma unroll
        for (int j = 0; j < BLK_N; ++j)
            acc[i][j] = (f32x8){};

    auto mma = [&](const bf16x16* a, const bf16x16* b) {
#pragma unroll
        for (int i = 0; i < BLK_M; ++i)
#pragma unroll
            for (int j = 0; j < BLK_N; ++j)
                acc[i][j] = __builtin_amdgcn_wmma_f32_16x16x32_bf16(
                                false, a[i], false, b[j], (short)0, acc[i][j],
                                false, false);
    };

    bf16x16 a0[BLK_M], b0[BLK_N];
    bf16x16 a1[BLK_M], b1[BLK_N];
    loadFrags(0, a0, b0);

    int kk = 0;
    for (; kk + 64 < K; kk += 64) {                 // K % 64 == 0 at all sites
        loadFrags(kk + 32, a1, b1);
        mma(a0, b0);
        loadFrags(kk + 64, a0, b0);
        mma(a1, b1);
    }
    loadFrags(kk + 32, a1, b1);                     // final pair
    mma(a0, b0);
    mma(a1, b1);

    // ---------------- epilogue ----------------
#pragma unroll
    for (int j = 0; j < BLK_N; ++j) {
        const int col = n0 + 16 * j + l16;
        const float bcol = (bias != nullptr && col < Nlog) ? bias[col] : 0.f;
#pragma unroll
        for (int i = 0; i < BLK_M; ++i) {
            const int rowBase = m0 + 16 * i + half * 8;
            if (outMode == OUT_BF16_TRANS) {
                // outT[col][rowBase..rowBase+7] -> one 16B store per lane.
                bf16x8 o;
#pragma unroll
                for (int r = 0; r < 8; ++r) {
                    float v = acc[i][j][r] + bcol;
                    if (doRelu) v = fmaxf(v, 0.f);
                    o[r] = f2bf(v);
                }
                *(bf16x8*)((__bf16*)out + (size_t)col * ldout + rowBase) = o;
            } else if (outMode == OUT_BF16_ROW) {
                __bf16* po = (__bf16*)out;
#pragma unroll
                for (int r = 0; r < 8; ++r) {
                    int row = rowBase + r;
                    float v = acc[i][j][r] + bcol;
                    if (maskb != nullptr) {
                        unsigned w = maskb[(size_t)row * MASK_WORDS + (col >> 5)];
                        if (!((w >> (col & 31)) & 1u)) v = 0.f;
                    }
                    if (doRelu) v = fmaxf(v, 0.f);
                    po[(size_t)row * ldout + col] = f2bf(v);
                }
            } else { // OUT_F32_ROW
                float* po = (float*)out;
                if (col < Nlog) {
#pragma unroll
                    for (int r = 0; r < 8; ++r) {
                        int row = rowBase + r;
                        float v = acc[i][j][r] + bcol;
                        if (doRelu) v = fmaxf(v, 0.f);
                        po[(size_t)row * ldout + col] = v;
                    }
                }
            }
        }
    }
}

// ---------------------------------------------------------------------------
// Row softmax over 21 classes. One thread per row.
// ---------------------------------------------------------------------------
__global__ __launch_bounds__(256)
void softmax21_kernel(const float* __restrict__ score, float* __restrict__ prob)
{
    int row = blockIdx.x * 256 + threadIdx.x;
    if (row >= N_ROIS) return;
    const float* s = score + (size_t)row * NCLS;
    float m = -1e30f;
    for (int cI = 0; cI < NCLS; ++cI) m = fmaxf(m, s[cI]);
    float e[NCLS];
    float sum = 0.f;
    for (int cI = 0; cI < NCLS; ++cI) { e[cI] = __expf(s[cI] - m); sum += e[cI]; }
    float inv = 1.f / sum;
    float* o = prob + (size_t)row * NCLS;
    for (int cI = 0; cI < NCLS; ++cI) o[cI] = e[cI] * inv;
}

// ---------------------------------------------------------------------------
// Host launcher
// ---------------------------------------------------------------------------
static inline int gemm_blocks(int M, int Npad) {
    int tiles = (M / (16 * BLK_M)) * (Npad / (16 * BLK_N));
    return (tiles + 7) / 8;                        // 8 waves (256 thr) / block
}

extern "C" void kernel_launch(void* const* d_in, const int* in_sizes, int n_in,
                              void* d_out, int out_size, void* d_ws, size_t ws_size,
                              hipStream_t stream) {
    (void)in_sizes; (void)n_in; (void)out_size; (void)ws_size;

    const float* rois   = (const float*)d_in[0];
    const float* pooled = (const float*)d_in[1];
    const float* W1     = (const float*)d_in[2];
    const float* b1     = (const float*)d_in[3];
    const float* W2     = (const float*)d_in[4];
    const float* b2     = (const float*)d_in[5];
    const float* Wc     = (const float*)d_in[6];
    const float* bc     = (const float*)d_in[7];
    const float* Wb     = (const float*)d_in[8];
    const float* bb     = (const float*)d_in[9];

    float* out       = (float*)d_out;
    float* out_cls   = out + (size_t)N_ROIS * 5;                       // 10240
    float* out_bbox  = out_cls + (size_t)N_ROIS * NCLS;                // +43008

    // Workspace layout (256B aligned), with buffer reuse:
    char* ws = (char*)d_ws;
    size_t off = 0;
    auto take = [&](size_t bytes) -> void* {
        void* p = ws + off;
        off += (bytes + 255) & ~(size_t)255;
        return p;
    };
    __bf16* bufA   = (__bf16*)take((size_t)N_ROIS * D_FEAT * 2);  // pooled_bf, later H2
    __bf16* bufB   = (__bf16*)take((size_t)N_ROIS * D_FEAT * 2);  // fn_bf, later Y2T
    __bf16* adj    = (__bf16*)take((size_t)N_ROIS * N_ROIS * 2);
    __bf16* W1T    = (__bf16*)take((size_t)H_DIM * D_FEAT * 2);
    __bf16* W2T    = (__bf16*)take((size_t)D_FEAT * H_DIM * 2);
    __bf16* WbT    = (__bf16*)take((size_t)NBB_PAD * D_FEAT * 2);
    __bf16* WcT    = (__bf16*)take((size_t)NCLS_PAD * D_FEAT * 2);
    __bf16* Y1T    = (__bf16*)take((size_t)H_DIM * N_ROIS * 2);
    __bf16* H1     = (__bf16*)take((size_t)N_ROIS * H_DIM * 2);
    unsigned* maskb = (unsigned*)take((size_t)N_ROIS * MASK_WORDS * 4);
    float* clsScore = (float*)take((size_t)N_ROIS * NCLS * 4);

    // rois pass-through (graph-capture safe async d2d copy)
    hipMemcpyAsync(out, rois, (size_t)N_ROIS * 5 * sizeof(float),
                   hipMemcpyDeviceToDevice, stream);

    // Stage 0: masks, norms, weight transposes (all independent)
    iou_mask_kernel<<<N_ROIS / 256, 256, 0, stream>>>(rois, maskb);
    norm_convert_kernel<<<N_ROIS, 256, 0, stream>>>(pooled, bufA, bufB);
    transpose_convert_kernel<<<2048, 256, 0, stream>>>(W1, W1T, D_FEAT, H_DIM, H_DIM);
    transpose_convert_kernel<<<2048, 256, 0, stream>>>(W2, W2T, H_DIM, D_FEAT, D_FEAT);
    transpose_convert_kernel<<<1024, 256, 0, stream>>>(Wb, WbT, D_FEAT, NBB, NBB_PAD);
    transpose_convert_kernel<<<512,  256, 0, stream>>>(Wc, WcT, D_FEAT, NCLS, NCLS_PAD);

    // adj = mask ? (fn @ fn^T) : 0           [2048 x 2048], K=4096
    gemm_wmma_bf16<<<gemm_blocks(N_ROIS, N_ROIS), 256, 0, stream>>>(
        bufB, D_FEAT, bufB, D_FEAT, adj, N_ROIS,
        N_ROIS, N_ROIS, N_ROIS, D_FEAT, nullptr, 0, maskb, OUT_BF16_ROW);

    // Y1^T = (pooled @ W1)^T                 [1024 x 2048], K=4096
    gemm_wmma_bf16<<<gemm_blocks(N_ROIS, H_DIM), 256, 0, stream>>>(
        bufA, D_FEAT, W1T, D_FEAT, Y1T, N_ROIS,
        N_ROIS, H_DIM, H_DIM, D_FEAT, nullptr, 0, nullptr, OUT_BF16_TRANS);

    // H1 = relu(adj @ Y1 + b1)               [2048 x 1024], K=2048
    gemm_wmma_bf16<<<gemm_blocks(N_ROIS, H_DIM), 256, 0, stream>>>(
        adj, N_ROIS, Y1T, N_ROIS, H1, H_DIM,
        N_ROIS, H_DIM, H_DIM, N_ROIS, b1, 1, nullptr, OUT_BF16_ROW);

    // Y2^T = (H1 @ W2)^T  (reuse fn buffer)  [4096 x 2048], K=1024
    gemm_wmma_bf16<<<gemm_blocks(N_ROIS, D_FEAT), 256, 0, stream>>>(
        H1, H_DIM, W2T, H_DIM, bufB, N_ROIS,
        N_ROIS, D_FEAT, D_FEAT, H_DIM, nullptr, 0, nullptr, OUT_BF16_TRANS);

    // H2 = relu(adj @ Y2 + b2) (reuse pooled buffer)  [2048 x 4096], K=2048
    gemm_wmma_bf16<<<gemm_blocks(N_ROIS, D_FEAT), 256, 0, stream>>>(
        adj, N_ROIS, bufB, N_ROIS, bufA, D_FEAT,
        N_ROIS, D_FEAT, D_FEAT, N_ROIS, b2, 1, nullptr, OUT_BF16_ROW);

    // bbox_pred = H2 @ W_bbox + b_bbox  -> f32 output   [2048 x 84], K=4096
    gemm_wmma_bf16<<<gemm_blocks(N_ROIS, NBB_PAD), 256, 0, stream>>>(
        bufA, D_FEAT, WbT, D_FEAT, out_bbox, NBB,
        N_ROIS, NBB, NBB_PAD, D_FEAT, bb, 0, nullptr, OUT_F32_ROW);

    // cls_score = H2 @ W_cls + b_cls                    [2048 x 21], K=4096
    gemm_wmma_bf16<<<gemm_blocks(N_ROIS, NCLS_PAD), 256, 0, stream>>>(
        bufA, D_FEAT, WcT, D_FEAT, clsScore, NCLS,
        N_ROIS, NCLS, NCLS_PAD, D_FEAT, bc, 0, nullptr, OUT_F32_ROW);

    // cls_prob = softmax(cls_score)
    softmax21_kernel<<<N_ROIS / 256, 256, 0, stream>>>(clsScore, out_cls);
}
